// DGN_63660005261335
// MI455X (gfx1250) — compile-verified, matrix operand order checked by reading, not run on hardware
//
#include <hip/hip_runtime.h>
#include <hip/hip_bf16.h>
#include <cstdint>
#include <math.h>

typedef __attribute__((ext_vector_type(16))) _Float16 v16h;
typedef __attribute__((ext_vector_type(8)))  float    v8f;

#define CTX    60
#define FACT   20
#define EDIM   100
#define HDIM   100
#define DESC   60
#define VOCAB  100000

// padded attention GEMM dims: att[MP][NP] = zf[MP][KP] @ W1h[NP][KP]^T
#define MP 64     // 60 rows padded
#define NP 112    // 100 cols padded
#define KP 224    // 200 inner padded
#define NBLK_BIG 391
#define PART_STRIDE 400

// ---------------------------------------------------------------- facts ----
__global__ void k_facts(const int* __restrict__ context,
                        const float* __restrict__ E_ctx,
                        float* __restrict__ facts) {
  int i = blockIdx.x;          // fact row 0..59
  int n = threadIdx.x;         // embed dim
  if (n >= EDIM) return;
  float en = (float)n / 99.0f;
  float acc = 0.f;
  for (int s = 0; s < FACT; ++s) {
    float sn = (float)s / 19.0f;
    float l = 1.0f - sn - en * (1.0f - 2.0f * sn);
    int tok = context[i * FACT + s];
    acc += E_ctx[(long long)tok * EDIM + n] * l;
  }
  facts[i * EDIM + n] = acc;
}

// ----------------------------------------------------------------- init ----
__global__ void k_init(const float* __restrict__ facts,
                       const float* __restrict__ Winit,
                       const float* __restrict__ binit,
                       const float* __restrict__ E_dec,
                       const float* __restrict__ W1,
                       float* __restrict__ m_ws, float* __restrict__ h_ws,
                       float* __restrict__ emb_ws, _Float16* __restrict__ W1h) {
  int tid = threadIdx.x;
  // zero-padded f16 copy of W1 (layout [NP][KP])
  for (int idx = tid; idx < NP * KP; idx += blockDim.x) {
    int n = idx / KP, k = idx % KP;
    float v = (n < HDIM && k < 2 * HDIM) ? W1[n * 2 * HDIM + k] : 0.f;
    W1h[idx] = (_Float16)v;
  }
  if (tid < HDIM) {
    float acc = binit[tid];
    const float* wr = Winit + (long long)tid * (CTX * HDIM);
    for (int k = 0; k < CTX * HDIM; ++k) acc += facts[k] * wr[k];
    m_ws[tid]   = fmaxf(acc, 0.f);
    h_ws[tid]   = 0.f;
    emb_ws[tid] = E_dec[1 * EDIM + tid];   // <SOS> token id 1
  }
}

// ----------------------------------------------------- recurrent (small) ---
__global__ void k_step_small(
    int t, const int* __restrict__ desc,
    const float* __restrict__ facts_g,
    const float* __restrict__ b1, const float* __restrict__ W2,
    const float* __restrict__ b2,
    const float* __restrict__ W3, const float* __restrict__ b3,
    const float* __restrict__ Wih, const float* __restrict__ Whh,
    const float* __restrict__ bih, const float* __restrict__ bhh,
    const float* __restrict__ Wo1, const float* __restrict__ bo1,
    const float* __restrict__ E_dec, const _Float16* __restrict__ W1h_g,
    float* __restrict__ m_ws, float* __restrict__ h_ws,
    float* __restrict__ emb_ws, float* __restrict__ u_ws,
    float* __restrict__ otok_ws) {
  extern __shared__ char smem[];
  float*    att   = (float*)smem;              // MP*NP
  float*    facts = att + MP * NP;             // CTX*HDIM
  float*    gi    = facts + CTX * HDIM;        // 3H
  float*    gh    = gi + 3 * HDIM;             // 3H
  float*    hN    = gh + 3 * HDIM;             // H (h_new)
  float*    cv    = hN + HDIM;                 // H (context c)
  float*    mP    = cv + HDIM;                 // H (m prev)
  float*    hP    = mP + HDIM;                 // H (h prev)
  float*    embL  = hP + HDIM;                 // H (emb prev)
  float*    gv    = embL + HDIM;               // 64 (gates)
  _Float16* zf    = (_Float16*)(gv + 64);      // MP*KP
  _Float16* w1s   = zf + MP * KP;              // NP*KP

  const int tid = threadIdx.x;
  const int NT  = blockDim.x;                  // 256 = 8 waves

  for (int idx = tid; idx < CTX * HDIM; idx += NT) facts[idx] = facts_g[idx];
  for (int idx = tid; idx < (NP * KP) / 2; idx += NT)
    ((uint32_t*)w1s)[idx] = ((const uint32_t*)W1h_g)[idx];
  if (tid < HDIM) { mP[tid] = m_ws[tid]; hP[tid] = h_ws[tid]; embL[tid] = emb_ws[tid]; }
  __syncthreads();

  // ---- GRU gates: gi = [emb,m] @ Wih^T + bih ; gh = h @ Whh^T + bhh
  for (int r = tid; r < 3 * HDIM; r += NT) {
    const float* wr = Wih + r * (EDIM + HDIM);
    float a = bih[r];
    for (int k = 0; k < EDIM; ++k) a += embL[k] * wr[k];
    for (int k = 0; k < HDIM; ++k) a += mP[k] * wr[EDIM + k];
    gi[r] = a;
    const float* wh = Whh + r * HDIM;
    float b = bhh[r];
    for (int k = 0; k < HDIM; ++k) b += hP[k] * wh[k];
    gh[r] = b;
  }
  __syncthreads();
  if (tid < HDIM) {
    float rr = 1.f / (1.f + expf(-(gi[tid] + gh[tid])));
    float zz = 1.f / (1.f + expf(-(gi[HDIM + tid] + gh[HDIM + tid])));
    float nn = tanhf(gi[2 * HDIM + tid] + rr * gh[2 * HDIM + tid]);
    hN[tid] = (1.f - zz) * nn + zz * hP[tid];
  }
  __syncthreads();

  // ---- zfeat = [|facts-h_new| , |facts-m|], zero-padded, f16
  for (int idx = tid; idx < MP * KP; idx += NT) {
    int i = idx / KP, k = idx % KP;
    float v = 0.f;
    if (i < CTX && k < 2 * HDIM)
      v = (k < HDIM) ? fabsf(facts[i * HDIM + k] - hN[k])
                     : fabsf(facts[i * HDIM + (k - HDIM)] - mP[k - HDIM]);
    zf[idx] = (_Float16)v;
  }
  __syncthreads();

  // ---- WMMA GEMM: att = zf @ w1s^T  (f16 in, f32 acc, 16x16x32)
  {
    const int lane = tid & 31, wave = tid >> 5;
    const int row16 = lane & 15;               // M (for A) / N (for B)
    const int kbase = (lane >> 4) * 8;         // K offset by lane half
    for (int tile = wave; tile < (MP / 16) * (NP / 16); tile += (NT >> 5)) {
      int tm = tile / (NP / 16);
      int tn = tile % (NP / 16);
      union { v8f v; float f[8]; } acc;
      #pragma unroll
      for (int r = 0; r < 8; ++r) acc.f[r] = 0.f;
      for (int kk = 0; kk < KP / 32; ++kk) {
        union { v16h v; _Float16 e[16]; } A, B;
        #pragma unroll
        for (int e = 0; e < 16; ++e) {
          int vv = e >> 1, hh = e & 1;
          int kloc = ((vv < 4) ? 0 : 16) + (vv & 3) * 2 + kbase + hh;
          A.e[e] = zf [(tm * 16 + row16) * KP + kk * 32 + kloc];
          B.e[e] = w1s[(tn * 16 + row16) * KP + kk * 32 + kloc];
        }
        acc.v = __builtin_amdgcn_wmma_f32_16x16x32_f16(
            false, A.v, false, B.v, (short)0, acc.v, false, false);
      }
      #pragma unroll
      for (int r = 0; r < 8; ++r) {
        int mr = r + 8 * (lane >> 4);
        att[(tm * 16 + mr) * NP + tn * 16 + (lane & 15)] = acc.f[r];
      }
    }
  }
  __syncthreads();

  // ---- gate g, softmax, context c
  if (tid < CTX) {
    float a = b2[0];
    for (int n = 0; n < HDIM; ++n) a += tanhf(att[tid * NP + n] + b1[n]) * W2[n];
    gv[tid] = a;
  }
  __syncthreads();
  if (tid == 0) {
    float mx = gv[0];
    for (int i = 1; i < CTX; ++i) mx = fmaxf(mx, gv[i]);
    float s = 0.f;
    for (int i = 0; i < CTX; ++i) { gv[i] = expf(gv[i] - mx); s += gv[i]; }
    float inv = 1.f / s;
    for (int i = 0; i < CTX; ++i) gv[i] *= inv;
  }
  __syncthreads();
  if (tid < HDIM) {
    float a = 0.f;
    for (int i = 0; i < CTX; ++i) a += gv[i] * facts[i * HDIM + tid];
    cv[tid] = a;
  }
  __syncthreads();

  // ---- m_new = relu([m,c,h] @ W3^T + b3) ; u = tanh([h,c] @ Wo1^T + bo1)
  if (tid < HDIM) {
    const float* w3r = W3 + tid * 3 * HDIM;
    float a = b3[tid];
    for (int k = 0; k < HDIM; ++k) a += mP[k] * w3r[k];
    for (int k = 0; k < HDIM; ++k) a += cv[k] * w3r[HDIM + k];
    for (int k = 0; k < HDIM; ++k) a += hN[k] * w3r[2 * HDIM + k];
    m_ws[tid] = fmaxf(a, 0.f);

    const float* wo = Wo1 + tid * 2 * HDIM;
    float b = bo1[tid];
    for (int k = 0; k < HDIM; ++k) b += hN[k] * wo[k];
    for (int k = 0; k < HDIM; ++k) b += cv[k] * wo[HDIM + k];
    u_ws[tid] = tanhf(b);

    h_ws[tid] = hN[tid];
    int tok = desc[t];
    emb_ws[tid] = E_dec[(long long)tok * EDIM + tid];  // teacher forcing
  }
  if (tid == 0) otok_ws[t] = 0.f;
}

// ------------------------------------------------- vocab pass (L2-bound) ---
__global__ void k_step_big(int t, const int* __restrict__ desc,
                           const float* __restrict__ u_ws,
                           const float* __restrict__ Wo2,
                           const float* __restrict__ bo2,
                           float* __restrict__ partial,
                           float* __restrict__ otok_ws) {
  __shared__ float us[HDIM];
  __shared__ float red[256];
  int tid = threadIdx.x;
  if (tid < HDIM) us[tid] = u_ws[tid];
  __syncthreads();

  long long v = (long long)blockIdx.x * blockDim.x + tid;
  float e = 0.f;
  if (v < VOCAB) {
    const float* row = Wo2 + v * HDIM;         // rows are 400B, 16B-aligned
    __builtin_prefetch(row, 0, 0);             // global_prefetch_b8
    float acc = bo2[v];
    const float4* r4 = (const float4*)row;
    #pragma unroll
    for (int q = 0; q < 25; ++q) {             // 25 * float4 = 100 floats
      float4 w = r4[q];
      acc += w.x * us[4 * q] + w.y * us[4 * q + 1] +
             w.z * us[4 * q + 2] + w.w * us[4 * q + 3];
    }
    e = expf(acc);
    if (v == desc[t]) otok_ws[t] = acc;
  }
  red[tid] = e;
  __syncthreads();
  for (int s = 128; s > 0; s >>= 1) {
    if (tid < s) red[tid] += red[tid + s];
    __syncthreads();
  }
  if (tid == 0) partial[t * PART_STRIDE + blockIdx.x] = red[0];
}

// ----------------------------------------------------------------- loss ----
__global__ void k_final(const float* __restrict__ partial,
                        const float* __restrict__ otok,
                        float* __restrict__ out) {
  float loss = 0.f;
  for (int t = 0; t < DESC; ++t) {
    float s = 0.f;
    for (int b = 0; b < NBLK_BIG; ++b) s += partial[t * PART_STRIDE + b];
    loss += logf(s) - otok[t];
  }
  out[0] = loss;
}

// --------------------------------------------------------------- launch ----
extern "C" void kernel_launch(void* const* d_in, const int* in_sizes, int n_in,
                              void* d_out, int out_size, void* d_ws, size_t ws_size,
                              hipStream_t stream) {
  const int*   context = (const int*)d_in[0];
  /* d_in[1] = fact_lengths (unused by the reference math) */
  const int*   desc  = (const int*)d_in[2];
  const float* E_ctx = (const float*)d_in[3];
  const float* E_dec = (const float*)d_in[4];
  const float* W1    = (const float*)d_in[5];
  const float* b1    = (const float*)d_in[6];
  const float* W2    = (const float*)d_in[7];
  const float* b2    = (const float*)d_in[8];
  const float* W3    = (const float*)d_in[9];
  const float* b3    = (const float*)d_in[10];
  const float* Wih   = (const float*)d_in[11];
  const float* Whh   = (const float*)d_in[12];
  const float* bih   = (const float*)d_in[13];
  const float* bhh   = (const float*)d_in[14];
  const float* Winit = (const float*)d_in[15];
  const float* binit = (const float*)d_in[16];
  const float* Wo1   = (const float*)d_in[17];
  const float* bo1   = (const float*)d_in[18];
  const float* Wo2   = (const float*)d_in[19];
  const float* bo2   = (const float*)d_in[20];

  float* ws      = (float*)d_ws;
  float* facts   = ws;                        // 6000
  float* m_ws    = ws + 6000;                 // 100
  float* h_ws    = ws + 6100;                 // 100
  float* emb_ws  = ws + 6200;                 // 100
  float* u_ws    = ws + 6300;                 // 100
  float* otok    = ws + 6400;                 // 60 (+4 pad)
  float* partial = ws + 6464;                 // 60*400
  _Float16* W1h  = (_Float16*)(ws + 6464 + DESC * PART_STRIDE); // NP*KP halves

  k_facts<<<CTX, 128, 0, stream>>>(context, E_ctx, facts);
  k_init <<<1, 256, 0, stream>>>(facts, Winit, binit, E_dec, W1,
                                 m_ws, h_ws, emb_ws, W1h);

  size_t smem = (size_t)(MP * NP + CTX * HDIM + 6 * HDIM + 5 * HDIM + 64) * sizeof(float)
              + (size_t)(MP * KP + NP * KP) * sizeof(_Float16);

  for (int t = 0; t < DESC; ++t) {
    k_step_small<<<1, 256, smem, stream>>>(
        t, desc, facts, b1, W2, b2, W3, b3, Wih, Whh, bih, bhh,
        Wo1, bo1, E_dec, W1h, m_ws, h_ws, emb_ws, u_ws, otok);
    k_step_big<<<NBLK_BIG, 256, 0, stream>>>(t, desc, u_ws, Wo2, bo2,
                                             partial, otok);
  }
  k_final<<<1, 1, 0, stream>>>(partial, otok, (float*)d_out);
}
